// SimpleSpikingNeuralNetwork_52329881534762
// MI455X (gfx1250) — matrix operand order repaired and checked by reading, not compiled, not run on hardware
//
#include <hip/hip_runtime.h>
#include <hip/hip_bf16.h>

#define B_N   1024
#define T_N   16
#define IN_N  2312
#define HID_N 1024
#define OUT_N 10
#define M_N   (B_N * T_N)   // 16384

typedef __attribute__((ext_vector_type(16))) __bf16 v16bf;
typedef __attribute__((ext_vector_type(8)))  __bf16 v8bf;
typedef __attribute__((ext_vector_type(8)))  float  v8f;
typedef __attribute__((ext_vector_type(2)))  float  v2f;

static __device__ inline v16bf cat16(v8bf a, v8bf b) {
  return __builtin_shufflevector(a, b, 0,1,2,3,4,5,6,7,8,9,10,11,12,13,14,15);
}

static __device__ inline float sigmoid_surr(float v) {
  // sigmoid(10*(v-1))
  return 1.0f / (1.0f + __expf(-10.0f * (v - 1.0f)));
}

// ---------------------------------------------------------------------------
// Kernel 1: h_all[m, n] = sum_k x[m,k] * W1[n,k] + b1[n]
// bf16x3 split GEMM, WMMA f32_16x16x32_bf16, tile 128(M) x 256(N), Kstep 32.
// Software pipeline: global loads for K-step k+1 are register-staged while
// the 48 WMMAs of step k run on the XDL pipe; global_prefetch_b8 two steps
// ahead keeps lines resident in L2/WGP$.
// grid = (HID/256, M/128), block = 256 threads (8 waves, 2x4 wave grid).
// ---------------------------------------------------------------------------
__global__ __launch_bounds__(256) void snn_gemm1(
    const float* __restrict__ x, const float* __restrict__ W1,
    const float* __restrict__ b1, float* __restrict__ h) {
  __shared__ __bf16 lAhi[128 * 32];
  __shared__ __bf16 lAlo[128 * 32];
  __shared__ __bf16 lBhi[32 * 256];
  __shared__ __bf16 lBlo[32 * 256];

  const int tid  = threadIdx.x;
  const int lane = tid & 31;
  const int wid  = tid >> 5;
  const int wm   = wid & 1;    // 0..1 -> M half (64 rows)
  const int wn   = wid >> 1;   // 0..3 -> N quarter (64 cols)
  const int m0   = blockIdx.y * 128;
  const int n0   = blockIdx.x * 256;

  v8f acc[4][4];
#pragma unroll
  for (int i = 0; i < 4; ++i)
#pragma unroll
    for (int j = 0; j < 4; ++j)
#pragma unroll
      for (int e = 0; e < 8; ++e) acc[i][j][e] = 0.0f;

  const int arow = tid >> 1;          // 0..127
  const int acol = (tid & 1) * 16;    // 0 / 16
  const int brow = tid;               // n-index 0..255

  const float* aPtr = x  + (size_t)(m0 + arow) * IN_N + acol; // + k gives elem
  const float* bPtr = W1 + (size_t)(n0 + brow) * IN_N;

  const int kh  = (lane >> 4) * 8;    // A-frag K sub-offset per half-lane
  const int mfl = lane & 15;

  float4 ar[4];   // staged A chunk: 16 floats (k = acol..acol+15)
  float4 br[8];   // staged B chunk: 32 floats (k = 0..31 of tile)

  // preload K-step 0 (always full: IN_N >= 32)
  {
    const float4* s4 = (const float4*)(aPtr);
#pragma unroll
    for (int q = 0; q < 4; ++q) ar[q] = s4[q];
    const float4* t4 = (const float4*)(bPtr);
#pragma unroll
    for (int q = 0; q < 8; ++q) br[q] = t4[q];
    __builtin_prefetch(aPtr + 32, 0, 1);
    __builtin_prefetch(bPtr + 32, 0, 1);
  }

  for (int k0 = 0; k0 < IN_N; k0 += 32) {
    // ---- stage current registers into LDS (fp32 -> bf16 hi/lo split) ----
#pragma unroll
    for (int q = 0; q < 4; ++q) {
      const float* v4 = (const float*)&ar[q];
#pragma unroll
      for (int e = 0; e < 4; ++e) {
        float v = v4[e];
        __bf16 hi = (__bf16)v;
        __bf16 lo = (__bf16)(v - (float)hi);
        lAhi[arow * 32 + acol + q * 4 + e] = hi;
        lAlo[arow * 32 + acol + q * 4 + e] = lo;
      }
    }
#pragma unroll
    for (int q = 0; q < 8; ++q) {
      const float* v4 = (const float*)&br[q];
#pragma unroll
      for (int e = 0; e < 4; ++e) {
        float v = v4[e];
        __bf16 hi = (__bf16)v;
        __bf16 lo = (__bf16)(v - (float)hi);
        lBhi[brow * 32 + q * 4 + e] = hi;
        lBlo[brow * 32 + q * 4 + e] = lo;
      }
    }
    __syncthreads();

    // ---- issue next tile's global loads + prefetch (overlap with WMMA) ----
    const int kn = k0 + 32;
    if (kn < IN_N) {
      if (kn + 32 <= IN_N) {
        const float4* s4 = (const float4*)(aPtr + kn);
#pragma unroll
        for (int q = 0; q < 4; ++q) ar[q] = s4[q];
        const float4* t4 = (const float4*)(bPtr + kn);
#pragma unroll
        for (int q = 0; q < 8; ++q) br[q] = t4[q];
        const int kp = (kn + 64 <= IN_N) ? kn + 32 : kn;
        __builtin_prefetch(aPtr + kp, 0, 1);
        __builtin_prefetch(bPtr + kp, 0, 1);
      } else {
        // ragged tail (K = 2312 = 72*32 + 8): zero-fill beyond K
#pragma unroll
        for (int q = 0; q < 4; ++q)
#pragma unroll
          for (int e = 0; e < 4; ++e) {
            int j = q * 4 + e;
            ((float*)&ar[q])[e] = (kn + acol + j < IN_N) ? aPtr[kn + j] : 0.0f;
          }
#pragma unroll
        for (int q = 0; q < 8; ++q)
#pragma unroll
          for (int e = 0; e < 4; ++e) {
            int j = q * 4 + e;
            ((float*)&br[q])[e] = (kn + j < IN_N) ? bPtr[kn + j] : 0.0f;
          }
      }
    }

    // ---- B fragments: lane = N col (lane%16); elems = 16 K values (kh..+7, kh+16..+23) ----
    v16bf bhi[4], blo[4];
#pragma unroll
    for (int in = 0; in < 4; ++in) {
      int nb = wn * 64 + in * 16;
      const __bf16* ph = &lBhi[(nb + mfl) * 32 + kh];
      const __bf16* pl = &lBlo[(nb + mfl) * 32 + kh];
      bhi[in] = cat16(*(const v8bf*)ph, *(const v8bf*)(ph + 16));
      blo[in] = cat16(*(const v8bf*)pl, *(const v8bf*)(pl + 16));
    }
#pragma unroll
    for (int im = 0; im < 4; ++im) {
      int mb = wm * 64 + im * 16 + mfl;
      const __bf16* ph = &lAhi[mb * 32 + kh];
      const __bf16* pl = &lAlo[mb * 32 + kh];
      v16bf ahi = cat16(*(const v8bf*)ph, *(const v8bf*)(ph + 16));
      v16bf alo = cat16(*(const v8bf*)pl, *(const v8bf*)(pl + 16));
#pragma unroll
      for (int in = 0; in < 4; ++in) {
        acc[im][in] = __builtin_amdgcn_wmma_f32_16x16x32_bf16(
            false, ahi, false, bhi[in], (short)0, acc[im][in], false, false);
        acc[im][in] = __builtin_amdgcn_wmma_f32_16x16x32_bf16(
            false, ahi, false, blo[in], (short)0, acc[im][in], false, false);
        acc[im][in] = __builtin_amdgcn_wmma_f32_16x16x32_bf16(
            false, alo, false, bhi[in], (short)0, acc[im][in], false, false);
      }
    }
    __syncthreads();
  }

  // ---- epilogue: C layout VGPR r -> M=r (lanes 0-15) / M=r+8 (lanes 16-31)
  const int nf = lane & 15;
  const int mh = lane >> 4;
#pragma unroll
  for (int im = 0; im < 4; ++im) {
#pragma unroll
    for (int in = 0; in < 4; ++in) {
      int mg = m0 + wm * 64 + im * 16;
      int ng = n0 + wn * 64 + in * 16 + nf;
      float bias = b1[ng];
#pragma unroll
      for (int r = 0; r < 8; ++r) {
        h[(size_t)(mg + r + 8 * mh) * HID_N + ng] = acc[im][in][r] + bias;
      }
    }
  }
}

// ---------------------------------------------------------------------------
// Kernel 2: temporal LIF scan. 8 batch rows per block; v1 in registers,
// s1 in LDS; layer-2 GEMM via v_wmma_f32_16x16x4_f32 (K split over 8 waves,
// cross-wave reduce with ds_add_f32 atomics).
// grid = B/8, block = 256 threads.
// ---------------------------------------------------------------------------
__global__ __launch_bounds__(256) void snn_temporal(
    const float* __restrict__ hws, const float* __restrict__ W2,
    const float* __restrict__ b2, float* __restrict__ out) {
  __shared__ float s1[8 * HID_N];   // 32 KB
  __shared__ float oacc[256];       // 16x16 padded output accumulator

  const int tid  = threadIdx.x;
  const int lane = tid & 31;
  const int wid  = tid >> 5;
  const int b0   = blockIdx.x * 8;

  // v1 ownership: row = wid (0..7), 32 consecutive hidden cols per thread
  const int vrow  = wid;
  const int vcol0 = lane * 32;
  float v1[32];
#pragma unroll
  for (int j = 0; j < 32; ++j) v1[j] = 0.0f;

  float v2 = 0.0f;                  // valid for tid < 80 (row=tid/10, col=tid%10)
  oacc[tid] = 0.0f;

  const int mrow = lane & 15;       // A-frag row
  const int kq   = (lane >> 4) * 2; // A-frag K sub-offset
  const int nf   = lane & 15;       // B-frag / C col

  for (int t = 0; t < T_N; ++t) {
    // ---- layer-1 LIF + spike into LDS ----
    const float* hp = hws + ((size_t)(b0 + vrow) * T_N + t) * HID_N + vcol0;
#pragma unroll
    for (int j = 0; j < 32; ++j) {
      float vn = 0.9f * v1[j] + hp[j];
      float sp = sigmoid_surr(vn);
      v1[j] = vn * (1.0f - sp);
      s1[vrow * HID_N + vcol0 + j] = sp;
    }
    __syncthreads();

    // ---- layer-2: o = s1 @ W2.T ; wave `wid` covers K in [wid*128, +128) ----
    v8f a2;
#pragma unroll
    for (int e = 0; e < 8; ++e) a2[e] = 0.0f;
#pragma unroll 4
    for (int kk = 0; kk < 128; kk += 4) {
      int kbase = wid * 128 + kk;
      v2f af;
      if (mrow < 8) {
        af.x = s1[mrow * HID_N + kbase + kq];
        af.y = s1[mrow * HID_N + kbase + kq + 1];
      } else {
        af.x = 0.0f; af.y = 0.0f;
      }
      v2f bf;
      int kb = kbase + 2 * (lane >> 4);
      bf.x = (nf < OUT_N) ? W2[nf * HID_N + kb]     : 0.0f;
      bf.y = (nf < OUT_N) ? W2[nf * HID_N + kb + 1] : 0.0f;
      a2 = __builtin_amdgcn_wmma_f32_16x16x4_f32(
          false, af, false, bf, (short)0, a2, false, false);
    }
#pragma unroll
    for (int r = 0; r < 8; ++r) {
      int m = r + 8 * (lane >> 4);
      atomicAdd(&oacc[m * 16 + nf], a2[r]);
    }
    __syncthreads();

    // ---- layer-2 LIF + output ----
    if (tid < 8 * OUT_N) {
      int row = tid / OUT_N, col = tid % OUT_N;
      float o  = oacc[row * 16 + col] + b2[col];
      float vn = 0.9f * v2 + o;
      float sp = sigmoid_surr(vn);
      v2 = vn * (1.0f - sp);
      out[((size_t)(b0 + row) * T_N + t) * OUT_N + col] = sp;
    }
    __syncthreads();
    oacc[tid] = 0.0f;   // completes before next iteration's post-sync atomics
  }
}

extern "C" void kernel_launch(void* const* d_in, const int* in_sizes, int n_in,
                              void* d_out, int out_size, void* d_ws, size_t ws_size,
                              hipStream_t stream) {
  const float* x  = (const float*)d_in[0];
  const float* W1 = (const float*)d_in[1];
  const float* b1 = (const float*)d_in[2];
  const float* W2 = (const float*)d_in[3];
  const float* b2 = (const float*)d_in[4];
  float* out = (float*)d_out;
  float* hws = (float*)d_ws;  // 16384 x 1024 fp32 = 64 MB

  dim3 g1(HID_N / 256, M_N / 128);
  snn_gemm1<<<g1, 256, 0, stream>>>(x, W1, b1, hws);
  snn_temporal<<<B_N / 8, 256, 0, stream>>>(hws, W2, b2, out);
}
